// FeatureInteraction_46480136077952
// MI455X (gfx1250) — matrix operand order repaired and checked by reading, not compiled, not run on hardware
//
#include <hip/hip_runtime.h>
#include <hip/hip_bf16.h>

typedef __attribute__((ext_vector_type(16))) _Float16     v16h;
typedef __attribute__((ext_vector_type(8)))  float        v8f;
typedef __attribute__((ext_vector_type(4)))  unsigned int u32x4;
typedef __attribute__((ext_vector_type(4)))  float        f32x4;

#define TS    5
#define PD    64
#define FD    33
#define IN_K  128        // 98 padded to 128
#define H1    32         // 2*ID
#define ID    16
#define MH    50
#define MHP   64         // 50 padded to 64
#define BPB   16         // batches per block
#define NW    4          // waves per block
#define NTHREADS (NW * 32)

__device__ __forceinline__ float silu_f(float x) {
    return x / (1.0f + __expf(-x));
}

// A-fragment (16x32 f16): lane holds row M=lane&15; halfs[0..15] = K k0+16*hi
// contiguous (matches B layout)
__device__ __forceinline__ v16h ldfragA(const u32x4* base, int strideU4, int row,
                                        int k0, int hi) {
    union { v16h h; u32x4 u[2]; } f;
    const u32x4* p = base + row * strideU4 + ((k0 >> 3) + hi * 2);
    f.u[0] = p[0];
    f.u[1] = p[1];
    return f.h;
}

// B-fragment (32x16 f16): lane holds col N=lane&15; halfs[0..15] = K k0+16*hi
// contiguous (same order as ldfragA). base points at W^T [N][Kpad] row-major.
__device__ __forceinline__ v16h ldfragB(const u32x4* base, int strideU4, int col,
                                        int k0, int hi) {
    union { v16h h; u32x4 u[2]; } f;
    const u32x4* p = base + col * strideU4 + ((k0 >> 3) + hi * 2);
    f.u[0] = p[0];
    f.u[1] = p[1];
    return f.h;
}

__global__ __launch_bounds__(NTHREADS)
void fi_kernel(const int* __restrict__ idsA, const int* __restrict__ idsB,
               const float* __restrict__ featsA, const float* __restrict__ featsB,
               const float* __restrict__ zA, const float* __restrict__ zB,
               const float* __restrict__ emb,
               const float* __restrict__ Wb1, const float* __restrict__ bb1,
               const float* __restrict__ Wb2, const float* __restrict__ bb2,
               const float* __restrict__ Wc1, const float* __restrict__ bc1,
               const float* __restrict__ Wc2, const float* __restrict__ bc2,
               const float* __restrict__ Wm1, const float* __restrict__ bm1,
               const float* __restrict__ Wm2, const float* __restrict__ bm2,
               const float* __restrict__ Watt, const float* __restrict__ batt,
               float* __restrict__ out, int Btot)
{
    __shared__ __align__(16) _Float16 sW1t[2][H1 * IN_K];  // W^T [32][128] blade/chest
    __shared__ __align__(16) _Float16 sW2t[2][ID * H1];    // W^T [16][32]
    __shared__ __align__(16) _Float16 sWm1t[MHP * H1];     // W^T [64][32] (K pad 16->32)
    __shared__ __align__(16) _Float16 sWattT[ID * H1];     // W^T [16][32]
    __shared__ float sb1[2][H1], sb2[2][ID];
    __shared__ float sbm1[MHP], sWm2p[MHP], sbatt[ID];
    __shared__ float sbm2v;
    __shared__ float sA[BPB * TS * ID];     // blade embeddings a
    __shared__ float sB[BPB * TS * ID];     // chest embeddings b
    __shared__ float sT[BPB * TS * ID];     // a @ Watt + batt
    __shared__ float sScore[BPB * 25];
    __shared__ float sLogit[BPB * 25];
    __shared__ __align__(16) _Float16 sX[NW][16 * IN_K];   // per-wave A staging

    const int tid   = threadIdx.x;
    const int wave  = tid >> 5;
    const int lane  = tid & 31;
    const int lrow  = lane & 15;
    const int hi    = lane >> 4;
    const int bbase = blockIdx.x * BPB;

    // ---------------- Phase 0: weights -> LDS (f32 -> f16, transpose, pad) ----
    for (int idx = tid; idx < H1 * IN_K; idx += NTHREADS) {
        const int n = idx >> 7, k = idx & 127;
        sW1t[0][idx] = (k < 98) ? (_Float16)Wb1[k * H1 + n] : (_Float16)0.0f;
        sW1t[1][idx] = (k < 98) ? (_Float16)Wc1[k * H1 + n] : (_Float16)0.0f;
    }
    for (int idx = tid; idx < ID * H1; idx += NTHREADS) {
        const int n = idx >> 5, k = idx & 31;
        sW2t[0][idx] = (_Float16)Wb2[k * ID + n];
        sW2t[1][idx] = (_Float16)Wc2[k * ID + n];
        sWattT[idx]  = (k < ID) ? (_Float16)Watt[k * ID + n] : (_Float16)0.0f;
    }
    for (int idx = tid; idx < MHP * H1; idx += NTHREADS) {
        const int n = idx >> 5, k = idx & 31;
        sWm1t[idx] = (n < MH && k < ID) ? (_Float16)Wm1[k * MH + n] : (_Float16)0.0f;
    }
    for (int i = tid; i < H1;  i += NTHREADS) { sb1[0][i] = bb1[i]; sb1[1][i] = bc1[i]; }
    for (int i = tid; i < ID;  i += NTHREADS) { sb2[0][i] = bb2[i]; sb2[1][i] = bc2[i]; sbatt[i] = batt[i]; }
    for (int i = tid; i < MHP; i += NTHREADS) {
        sbm1[i]  = (i < MH) ? bm1[i] : 0.0f;
        sWm2p[i] = (i < MH) ? Wm2[i] : 0.0f;
    }
    if (tid == 0) sbm2v = bm2[0];
    __syncthreads();

    _Float16* X  = &sX[wave][0];
    u32x4*    Xu = (u32x4*)X;

    // ---------------- Phase 1: blade/chest MLPs via WMMA --------------------
    // 2 teams x 80 rows = 10 row-tiles of 16; tiles 0..4 = team A, 5..9 = team B
    for (int t = wave; t < 2 * TS; t += NW) {
        const int team = (t < TS) ? 0 : 1;
        const int rb   = (team ? (t - TS) : t) * 16;

        // -- stage X tile [16 x 128] f16 (emb | feats | ctx | zero-pad) --
        {
            const int r  = lane >> 1;
            const int h  = lane & 1;
            const int gr = rb + r;
            const int bt = gr / TS, sl = gr % TS;
            int gb = bbase + bt; if (gb >= Btot) gb = Btot - 1;
            if (h == 0) {
                const int id = (team ? idsB : idsA)[gb * TS + sl];
                const f32x4* e = (const f32x4*)(emb + (size_t)id * PD);
                for (int q = 0; q < 8; ++q) {
                    f32x4 v0 = e[2 * q], v1 = e[2 * q + 1];
                    union { _Float16 hh[8]; u32x4 u; } pk;
                    pk.hh[0] = (_Float16)v0.x; pk.hh[1] = (_Float16)v0.y;
                    pk.hh[2] = (_Float16)v0.z; pk.hh[3] = (_Float16)v0.w;
                    pk.hh[4] = (_Float16)v1.x; pk.hh[5] = (_Float16)v1.y;
                    pk.hh[6] = (_Float16)v1.z; pk.hh[7] = (_Float16)v1.w;
                    Xu[r * 16 + q] = pk.u;
                }
            } else {
                const float* f = (team ? featsB : featsA) + ((size_t)gb * TS + sl) * FD;
                union { _Float16 hh[64]; u32x4 u[8]; } buf;
                for (int q = 0; q < FD; ++q) buf.hh[q] = (_Float16)f[q];
                buf.hh[FD] = (_Float16)((team ? zB : zA)[gb]);
                for (int q = FD + 1; q < 64; ++q) buf.hh[q] = (_Float16)0.0f;
                for (int q = 0; q < 8; ++q) Xu[r * 16 + 8 + q] = buf.u[q];
            }
        }

        // -- layer 1: [16x128] @ [128x32] : 4 k-steps, 2 n-tiles --
        const u32x4* W1u = (const u32x4*)&sW1t[team][0];
        v8f acc0 = {}; v8f acc1 = {};
        for (int k = 0; k < 4; ++k) {
            v16h af = ldfragA(Xu, 16, lrow, k * 32, hi);
            v16h b0 = ldfragB(W1u, 16, lrow, k * 32, hi);
            acc0 = __builtin_amdgcn_wmma_f32_16x16x32_f16(false, af, false, b0,
                                                          (short)0, acc0, false, false);
            v16h b1 = ldfragB(W1u, 16, 16 + lrow, k * 32, hi);
            acc1 = __builtin_amdgcn_wmma_f32_16x16x32_f16(false, af, false, b1,
                                                          (short)0, acc1, false, false);
        }

        // -- bias + SiLU in C-layout, restage as f16 A-tile [16 x 32] --
        {
            const float bn0 = sb1[team][lrow];
            const float bn1 = sb1[team][16 + lrow];
            for (int v = 0; v < 8; ++v) {
                const int M = v + hi * 8;
                X[M * 32 + lrow]      = (_Float16)silu_f(acc0[v] + bn0);
                X[M * 32 + 16 + lrow] = (_Float16)silu_f(acc1[v] + bn1);
            }
        }

        // -- layer 2: [16x32] @ [32x16] --
        v8f acc2 = {};
        {
            v16h af = ldfragA(Xu, 4, lrow, 0, hi);
            v16h bf = ldfragB((const u32x4*)&sW2t[team][0], 4, lrow, 0, hi);
            acc2 = __builtin_amdgcn_wmma_f32_16x16x32_f16(false, af, false, bf,
                                                          (short)0, acc2, false, false);
        }
        float* dst = team ? sB : sA;
        const float b2 = sb2[team][lrow];
        float aval[8];
        for (int v = 0; v < 8; ++v) {
            const int M = v + hi * 8;
            aval[v] = acc2[v] + b2;
            dst[(rb + M) * ID + lrow] = aval[v];
        }

        // -- attention projection (blade only): att = a @ Watt + batt --
        if (team == 0) {
            for (int v = 0; v < 8; ++v) {
                const int M = v + hi * 8;
                X[M * 32 + lrow]      = (_Float16)aval[v];
                X[M * 32 + 16 + lrow] = (_Float16)0.0f;
            }
            v8f acc3 = {};
            v16h af = ldfragA(Xu, 4, lrow, 0, hi);
            v16h bf = ldfragB((const u32x4*)sWattT, 4, lrow, 0, hi);
            acc3 = __builtin_amdgcn_wmma_f32_16x16x32_f16(false, af, false, bf,
                                                          (short)0, acc3, false, false);
            for (int v = 0; v < 8; ++v) {
                const int M = v + hi * 8;
                sT[(rb + M) * ID + lrow] = acc3[v] + sbatt[lrow];
            }
        }
    }
    __syncthreads();

    // ---------------- Phase 2: pairwise scoring MLP via WMMA ----------------
    // 16 batches x 25 pairs = 400 rows -> 25 row-tiles of 16
    for (int t = wave; t < 25; t += NW) {
        const int pr  = t * 16 + lrow;
        const int bt  = pr / 25;
        const int pix = pr % 25;
        const int ii  = pix / TS, jj = pix % TS;
        const float* av = &sA[(bt * TS + ii) * ID + hi * 8];
        const float* bv = &sB[(bt * TS + jj) * ID + hi * 8];
        const float* tv = &sT[(bt * TS + ii) * ID + hi * 8];

        // stage interact row (a_i * b_j) as f16 A-tile, K pad 16->32; logits too
        float part = 0.0f;
        union { _Float16 hh[8]; u32x4 u; } pk;
        for (int d = 0; d < 8; ++d) {
            const float bd = bv[d];
            pk.hh[d] = (_Float16)(av[d] * bd);
            part += tv[d] * bd;
        }
        Xu[lrow * 4 + hi] = pk.u;
        u32x4 zz = {0u, 0u, 0u, 0u};
        Xu[lrow * 4 + 2 + hi] = zz;
        part += __shfl_xor(part, 16, 32);
        if (hi == 0) sLogit[pr] = part;

        // hidden: [16x32] @ [32x64] (Wm1^T padded), then dot with Wm2
        v16h af = ldfragA(Xu, 4, lrow, 0, hi);
        float rowdot[8] = {0.f, 0.f, 0.f, 0.f, 0.f, 0.f, 0.f, 0.f};
        for (int n = 0; n < 4; ++n) {
            v16h bf = ldfragB((const u32x4*)sWm1t, 4, n * 16 + lrow, 0, hi);
            v8f acc = {};
            acc = __builtin_amdgcn_wmma_f32_16x16x32_f16(false, af, false, bf,
                                                         (short)0, acc, false, false);
            const int N     = n * 16 + lrow;
            const float bia = sbm1[N];
            const float w2  = sWm2p[N];
            for (int v = 0; v < 8; ++v) rowdot[v] += silu_f(acc[v] + bia) * w2;
        }
        for (int v = 0; v < 8; ++v)
            for (int m = 1; m < 16; m <<= 1)
                rowdot[v] += __shfl_xor(rowdot[v], m, 32);
        if (lrow == 0) {
            for (int v = 0; v < 8; ++v) {
                const int pr2 = t * 16 + v + hi * 8;
                sScore[pr2] = rowdot[v] + sbm2v;
            }
        }
    }
    __syncthreads();

    // ---------------- Phase 3: per-batch softmax + reduce -------------------
    if (tid < BPB) {
        const int gb = bbase + tid;
        const float* lg = &sLogit[tid * 25];
        const float* sc = &sScore[tid * 25];
        float total = 0.0f;
        for (int i = 0; i < TS; ++i) {
            float mx = -1e30f;
            for (int j = 0; j < TS; ++j) mx = fmaxf(mx, lg[i * TS + j]);
            float e[TS], den = 0.0f;
            for (int j = 0; j < TS; ++j) { e[j] = __expf(lg[i * TS + j] - mx); den += e[j]; }
            const float inv = 1.0f / den;
            for (int j = 0; j < TS; ++j) total += sc[i * TS + j] * e[j] * inv;
        }
        if (gb < Btot) out[gb] = total;
    }
}

extern "C" void kernel_launch(void* const* d_in, const int* in_sizes, int n_in,
                              void* d_out, int out_size, void* d_ws, size_t ws_size,
                              hipStream_t stream) {
    (void)n_in; (void)out_size; (void)d_ws; (void)ws_size;
    const int B = in_sizes[4];                 // z_g_A has shape (B,)
    dim3 grid((B + BPB - 1) / BPB), block(NTHREADS);
    fi_kernel<<<grid, block, 0, stream>>>(
        (const int*)d_in[0],  (const int*)d_in[1],
        (const float*)d_in[2], (const float*)d_in[3],
        (const float*)d_in[4], (const float*)d_in[5],
        (const float*)d_in[6],
        (const float*)d_in[7],  (const float*)d_in[8],
        (const float*)d_in[9],  (const float*)d_in[10],
        (const float*)d_in[11], (const float*)d_in[12],
        (const float*)d_in[13], (const float*)d_in[14],
        (const float*)d_in[15], (const float*)d_in[16],
        (const float*)d_in[17], (const float*)d_in[18],
        (const float*)d_in[19], (const float*)d_in[20],
        (float*)d_out, B);
}
